// TransformerBlock_5720896438291
// MI455X (gfx1250) — matrix operand order repaired and checked by reading, not compiled, not run on hardware
//
#include <hip/hip_runtime.h>
#include <hip/hip_bf16.h>

// ---------- vector types for WMMA fragments ----------
typedef __attribute__((ext_vector_type(16))) _Float16 v16h;
typedef __attribute__((ext_vector_type(8)))  _Float16 v8h;
typedef __attribute__((ext_vector_type(4)))  _Float16 v4h;
typedef __attribute__((ext_vector_type(8)))  float    v8f;
typedef __attribute__((ext_vector_type(8)))  unsigned short v8u;

#define WMMA_F16(a,b,c) \
  __builtin_amdgcn_wmma_f32_16x16x32_f16(false,(a),false,(b),(short)0,(c),false,false)

// DPP16 row_ror:N (rotate within 16-lane rows) -- pure-VALU butterfly shuffles
#define ROR16(x, N) \
  __int_as_float(__builtin_amdgcn_update_dpp(0, __float_as_int(x), 0x120 + (N), 0xf, 0xf, true))

static __device__ __forceinline__ float row_max16(float v) {
  v = fmaxf(v, ROR16(v, 1));
  v = fmaxf(v, ROR16(v, 2));
  v = fmaxf(v, ROR16(v, 4));
  v = fmaxf(v, ROR16(v, 8));
  return v;
}
static __device__ __forceinline__ float row_sum16(float v) {
  v += ROR16(v, 1);
  v += ROR16(v, 2);
  v += ROR16(v, 4);
  v += ROR16(v, 8);
  return v;
}

static __device__ __forceinline__ v16h frag_cat(v8h lo, v8h hi) {
  v16h r;
#pragma unroll
  for (int i = 0; i < 8; ++i) { r[i] = lo[i]; r[i + 8] = hi[i]; }
  return r;
}

// =====================================================================
// RMSNorm: f32 [rows, D=2048] -> f16 [rows, D], one block (256 thr) per row
// =====================================================================
__global__ __launch_bounds__(256)
void rmsnorm_f16_kernel(const float* __restrict__ X, const float* __restrict__ g,
                        _Float16* __restrict__ Y, int D) {
  __shared__ float red[256];
  const int tid = threadIdx.x;
  const size_t row = blockIdx.x;
  const float4* x4 = (const float4*)(X + row * (size_t)D);
  const float4* g4 = (const float4*)g;
  const int n4 = D >> 2;
  float ss = 0.f;
  for (int i = tid; i < n4; i += 256) {
    const float4 v = x4[i];
    ss += v.x * v.x + v.y * v.y + v.z * v.z + v.w * v.w;
  }
  red[tid] = ss;
  __syncthreads();
  for (int s = 128; s > 0; s >>= 1) {
    if (tid < s) red[tid] += red[tid + s];
    __syncthreads();
  }
  const float r = rsqrtf(red[0] / (float)D + 1.1920929e-07f);
  v4h* y4 = (v4h*)(Y + row * (size_t)D);
  for (int i = tid; i < n4; i += 256) {
    const float4 v = x4[i];
    const float4 gg = g4[i];
    v4h o = { (_Float16)(v.x * r * gg.x), (_Float16)(v.y * r * gg.y),
              (_Float16)(v.z * r * gg.z), (_Float16)(v.w * r * gg.w) };
    y4[i] = o;
  }
}

// =====================================================================
// GEMM: C[M,N] = A[M,K](f16) @ W[N,K]^T (f32, converted to f16 in LDS)
// 256 threads = 8 waves (2x4); block tile 128x128x32; wave tile 64x32.
// Double-buffered LDS, one barrier per K-step.
// EPI (compile-time): 1 = store f16, 2 = f32 + residual, 3 = GELU -> f16
// =====================================================================
#define BM 128
#define BN 128
#define BK 32
#define LDT 40   // padded LDS row stride in f16 (80B, keeps 16B alignment)

template <int EPI>
__global__ __launch_bounds__(256)
void gemm_f16_kernel(const _Float16* __restrict__ A, const float* __restrict__ B,
                     float* __restrict__ Cf, _Float16* __restrict__ Ch,
                     const float* __restrict__ Res,
                     int M, int N, int K) {
  __shared__ _Float16 sm[2][(BM + BN) * LDT];   // [buf][A tile | B tile]

  const int tid  = threadIdx.x;
  const int lane = tid & 31;
  const int wave = tid >> 5;
  const int wm = wave >> 2;      // 0..1
  const int wn = wave & 3;       // 0..3
  const int lr = lane & 15;
  const int lg = lane >> 4;
  const int m0 = blockIdx.y * BM;
  const int n0 = blockIdx.x * BN;
  const int NK = K / BK;

  v8f acc[4][2] = {};
  uint4  areg[2];
  float4 breg[4];

  // ---- global-load helper indices (A: 512x16B chunks, B: 1024x float4) ----
  const int arow0 = tid >> 2, aseg0 = (tid & 3) * 8;
  const int brow0 = tid >> 3, bseg0 = (tid & 7) * 4;

  // ---- prologue: stage tile 0 ----
#pragma unroll
  for (int i = 0; i < 2; ++i) {
    const int c = tid + 256 * i;
    areg[i] = *(const uint4*)(A + (size_t)(m0 + (c >> 2)) * K + (c & 3) * 8);
  }
#pragma unroll
  for (int i = 0; i < 4; ++i) {
    const int c = tid + 256 * i;
    breg[i] = *(const float4*)(B + (size_t)(n0 + (c >> 3)) * K + (c & 7) * 4);
  }
  {
    _Float16* At = &sm[0][0];
    _Float16* Bt = &sm[0][BM * LDT];
#pragma unroll
    for (int i = 0; i < 2; ++i) {
      const int c = tid + 256 * i;
      *(uint4*)&At[(c >> 2) * LDT + (c & 3) * 8] = areg[i];
    }
#pragma unroll
    for (int i = 0; i < 4; ++i) {
      const int c = tid + 256 * i;
      const float4 f = breg[i];
      v4h hv = { (_Float16)f.x, (_Float16)f.y, (_Float16)f.z, (_Float16)f.w };
      *(v4h*)&Bt[(c >> 3) * LDT + (c & 7) * 4] = hv;
    }
  }
  __syncthreads();

  for (int kt = 0; kt < NK; ++kt) {
    const int cur = kt & 1;
    // ---- issue global loads for next tile (overlap with compute) ----
    if (kt + 1 < NK) {
      const int k1 = (kt + 1) * BK;
#pragma unroll
      for (int i = 0; i < 2; ++i) {
        const int c = tid + 256 * i;
        areg[i] = *(const uint4*)(A + (size_t)(m0 + (c >> 2)) * K + k1 + (c & 3) * 8);
      }
#pragma unroll
      for (int i = 0; i < 4; ++i) {
        const int c = tid + 256 * i;
        breg[i] = *(const float4*)(B + (size_t)(n0 + (c >> 3)) * K + k1 + (c & 7) * 4);
      }
      // prefetch one more tile ahead (global_prefetch_b8)
      if (kt + 2 < NK) {
        __builtin_prefetch(A + (size_t)(m0 + arow0) * K + k1 + BK + aseg0, 0, 1);
        __builtin_prefetch(B + (size_t)(n0 + brow0) * K + k1 + BK + bseg0, 0, 1);
      }
    }

    // ---- compute from current buffer ----
    {
      const _Float16* At = &sm[cur][0];
      const _Float16* Bt = &sm[cur][BM * LDT];
      v16h af[4], bf[2];
#pragma unroll
      for (int mm = 0; mm < 4; ++mm) {
        const int ar = wm * 64 + mm * 16 + lr;
        v8h lo = *(const v8h*)&At[ar * LDT + lg * 8];       // K = lg*8 .. +7
        v8h hi = *(const v8h*)&At[ar * LDT + 16 + lg * 8];  // K = 16+lg*8 ..
        af[mm] = frag_cat(lo, hi);
      }
#pragma unroll
      for (int nn = 0; nn < 2; ++nn) {
        const int br = wn * 32 + nn * 16 + lr;
        v8h lo = *(const v8h*)&Bt[br * LDT + lg * 8];       // K = lg*8 .. +7
        v8h hi = *(const v8h*)&Bt[br * LDT + 16 + lg * 8];  // K = 16+lg*8 ..
        bf[nn] = frag_cat(lo, hi);
      }
#pragma unroll
      for (int mm = 0; mm < 4; ++mm)
#pragma unroll
        for (int nn = 0; nn < 2; ++nn)
          acc[mm][nn] = WMMA_F16(af[mm], bf[nn], acc[mm][nn]);
    }

    // ---- store staged regs into the other buffer ----
    if (kt + 1 < NK) {
      _Float16* At = &sm[1 - cur][0];
      _Float16* Bt = &sm[1 - cur][BM * LDT];
#pragma unroll
      for (int i = 0; i < 2; ++i) {
        const int c = tid + 256 * i;
        *(uint4*)&At[(c >> 2) * LDT + (c & 3) * 8] = areg[i];
      }
#pragma unroll
      for (int i = 0; i < 4; ++i) {
        const int c = tid + 256 * i;
        const float4 f = breg[i];
        v4h hv = { (_Float16)f.x, (_Float16)f.y, (_Float16)f.z, (_Float16)f.w };
        *(v4h*)&Bt[(c >> 3) * LDT + (c & 7) * 4] = hv;
      }
    }
    __syncthreads();
  }

  // ---- straight-line epilogue (C layout: m = v + 8*lg, n = lr) ----
  const int mbase = m0 + wm * 64 + 8 * lg;
  const int nbase = n0 + wn * 32 + lr;
#pragma unroll
  for (int v = 0; v < 8; ++v) {
    const size_t idx0 = (size_t)(mbase + v) * N + nbase;
#pragma unroll
    for (int mm = 0; mm < 4; ++mm) {
      const size_t idxm = idx0 + (size_t)(mm * 16) * N;
#pragma unroll
      for (int nn = 0; nn < 2; ++nn) {
        const size_t idx = idxm + nn * 16;
        const float x = acc[mm][nn][v];
        if (EPI == 1) {
          Ch[idx] = (_Float16)x;
        } else if (EPI == 2) {
          Cf[idx] = x + Res[idx];
        } else {  // EPI == 3: exact GELU
          Ch[idx] = (_Float16)(0.5f * x * (1.0f + erff(x * 0.70710678118654752f)));
        }
      }
    }
  }
}

// =====================================================================
// Flash attention: causal, H=16, Dh=128, f16 in/out, f32 online softmax.
// Block = 128 threads = 4 waves; each wave owns 16 q rows of a 64-row tile.
// Grid: (S/64, B*H).
// =====================================================================
#define LDK 136   // K tile stride (128 + 8 f16 pad)
#define LDV 40    // V^T tile stride (32 + 8)
#define LDP 40    // P tile stride

__global__ __launch_bounds__(128)
void attn_kernel(const _Float16* __restrict__ Q, const _Float16* __restrict__ Kx,
                 const _Float16* __restrict__ Vx, _Float16* __restrict__ O, int S) {
  __shared__ _Float16 Ks[32 * LDK];     // [kv=32][d=128]
  __shared__ _Float16 Vs[128 * LDV];    // V^T: [d=128][kv=32]
  __shared__ _Float16 Ps[4 * 16 * LDP]; // per-wave P tile [16][32]

  const int tid = threadIdx.x, lane = tid & 31, wave = tid >> 5;
  const int lr = lane & 15, lg = lane >> 4;
  const int bh = blockIdx.y, b = bh >> 4, h = bh & 15;
  const int q0 = blockIdx.x * 64;
  const int qw = q0 + wave * 16;
  const size_t rbase = ((size_t)b * S) * 2048 + (size_t)h * 128;

  // Q fragments resident in registers: 16 rows x 128 d (4 frags of K=32)
  v16h qf[4];
#pragma unroll
  for (int f = 0; f < 4; ++f) {
    const _Float16* qp = Q + rbase + (size_t)(qw + lr) * 2048 + f * 32 + lg * 8;
    v8h lo = *(const v8h*)qp;
    v8h hi = *(const v8h*)(qp + 16);
    qf[f] = frag_cat(lo, hi);
  }

  v8f o[8] = {};
  float mrun[8], lrun[8];
#pragma unroll
  for (int v = 0; v < 8; ++v) { mrun[v] = -3.0e38f; lrun[v] = 0.f; }
  const float sc = 0.08838834764831845f;   // 1/sqrt(128)

  const int steps = (q0 + 64) / 32;        // causal upper bound, uniform per block
  for (int jt = 0; jt < steps; ++jt) {
    const int j0 = jt * 32;
    // ---- stage K tile: 32 x 128 f16 (512 x 16B chunks / 128 thr) ----
#pragma unroll
    for (int i = 0; i < 4; ++i) {
      const int c = tid + 128 * i, row = c >> 4, seg = c & 15;
      *(uint4*)&Ks[row * LDK + seg * 8] =
          *(const uint4*)(Kx + rbase + (size_t)(j0 + row) * 2048 + seg * 8);
    }
    // ---- stage V^T tile: pair kv rows -> packed b32 stores ----
#pragma unroll
    for (int i = 0; i < 2; ++i) {
      const int c = tid + 128 * i;          // 0..255
      const int kv = (c >> 4) * 2;          // 0,2,..,30
      const int seg = c & 15;               // d0 = seg*8
      const _Float16* vp = Vx + rbase + (size_t)(j0 + kv) * 2048 + seg * 8;
      const v8u a = __builtin_bit_cast(v8u, *(const v8h*)vp);
      const v8u bq = __builtin_bit_cast(v8u, *(const v8h*)(vp + 2048));
#pragma unroll
      for (int e = 0; e < 8; ++e) {
        const unsigned u = (unsigned)a[e] | ((unsigned)bq[e] << 16);
        *(unsigned*)&Vs[(seg * 8 + e) * LDV + kv] = u;
      }
    }
    __syncthreads();

    // ---- scores: S = Q K^T  (two 16x16 n-tiles over 32 kv) ----
    v8f s0 = {}, s1 = {};
#pragma unroll
    for (int f = 0; f < 4; ++f) {
      v8h lo = *(const v8h*)&Ks[lr * LDK + f * 32 + lg * 8];
      v8h hi = *(const v8h*)&Ks[lr * LDK + f * 32 + 16 + lg * 8];
      s0 = WMMA_F16(qf[f], frag_cat(lo, hi), s0);
      lo = *(const v8h*)&Ks[(16 + lr) * LDK + f * 32 + lg * 8];
      hi = *(const v8h*)&Ks[(16 + lr) * LDK + f * 32 + 16 + lg * 8];
      s1 = WMMA_F16(qf[f], frag_cat(lo, hi), s1);
    }

    // ---- online softmax in C-layout (row m = v + 8*lg across lanes),
    //      DPP row_ror butterflies (no LDS traffic) ----
    _Float16* pw = &Ps[wave * 16 * LDP];
#pragma unroll
    for (int v = 0; v < 8; ++v) {
      const int qa = qw + v + 8 * lg;
      const float a0 = s0[v] * sc + ((j0 + lr)      > qa ? -1e9f : 0.f);
      const float a1 = s1[v] * sc + ((j0 + 16 + lr) > qa ? -1e9f : 0.f);
      const float mx = row_max16(fmaxf(a0, a1));
      const float mnew  = fmaxf(mrun[v], mx);
      const float scale = __expf(mrun[v] - mnew);
      const float p0 = __expf(a0 - mnew), p1 = __expf(a1 - mnew);
      const float rs = row_sum16(p0 + p1);
      lrun[v] = lrun[v] * scale + rs;
      mrun[v] = mnew;
#pragma unroll
      for (int d = 0; d < 8; ++d) o[d][v] *= scale;
      pw[(v + 8 * lg) * LDP + lr]      = (_Float16)p0;
      pw[(v + 8 * lg) * LDP + 16 + lr] = (_Float16)p1;
    }
    // P tile is produced and consumed by this wave only: DS ops are wave-wide
    // and in-order (DScnt), so no block barrier is needed here.

    // ---- O += P @ V  (A-frag from P tile, B-frags from V^T tile) ----
    v8h plo = *(const v8h*)&pw[lr * LDP + lg * 8];
    v8h phi = *(const v8h*)&pw[lr * LDP + 16 + lg * 8];
    v16h pf = frag_cat(plo, phi);
#pragma unroll
    for (int d = 0; d < 8; ++d) {
      v8h lo = *(const v8h*)&Vs[(d * 16 + lr) * LDV + lg * 8];
      v8h hi = *(const v8h*)&Vs[(d * 16 + lr) * LDV + 16 + lg * 8];
      o[d] = WMMA_F16(pf, frag_cat(lo, hi), o[d]);
    }
    __syncthreads();
  }

  // ---- normalize and store (base pointer + immediate offsets) ----
#pragma unroll
  for (int v = 0; v < 8; ++v) {
    _Float16* op = O + rbase + (size_t)(qw + v + 8 * lg) * 2048 + lr;
    const float inv = 1.0f / lrun[v];
#pragma unroll
    for (int d = 0; d < 8; ++d)
      op[d * 16] = (_Float16)(o[d][v] * inv);
  }
}

// =====================================================================
// Host-side orchestration
// =====================================================================
extern "C" void kernel_launch(void* const* d_in, const int* in_sizes, int n_in,
                              void* d_out, int out_size, void* d_ws, size_t ws_size,
                              hipStream_t stream) {
  const float* X      = (const float*)d_in[0];
  // d_in[1] = causal_mask: unused (mask applied analytically)
  const float* g_attn = (const float*)d_in[2];
  const float* g_ffn  = (const float*)d_in[3];
  const float* Wq = (const float*)d_in[4];
  const float* Wk = (const float*)d_in[5];
  const float* Wv = (const float*)d_in[6];
  const float* Wo = (const float*)d_in[7];
  const float* W1 = (const float*)d_in[8];
  const float* W2 = (const float*)d_in[9];
  float* out = (float*)d_out;

  const int Bz = 2, S = 2048, D = 2048, F = 8192;
  const int M = Bz * S;                    // 4096 token rows

  char* ws = (char*)d_ws;
  const size_t szMD_h = (size_t)M * D * sizeof(_Float16);   // 16.78 MB
  _Float16* Xn = (_Float16*)(ws + 0 * szMD_h);  // normalized activations (reused as hn)
  _Float16* Qb = (_Float16*)(ws + 1 * szMD_h);
  _Float16* Kb = (_Float16*)(ws + 2 * szMD_h);
  _Float16* Vb = (_Float16*)(ws + 3 * szMD_h);
  _Float16* AO = (_Float16*)(ws + 4 * szMD_h);
  _Float16* F1 = Qb;                            // 67.1 MB, aliases dead Q/K/V/AO
  float*    Hb = (float*)(ws + 5 * szMD_h);     // f32 residual h, 33.55 MB

  // 1) Xn = RMSNorm(X) * g_attn  (f16)
  rmsnorm_f16_kernel<<<M, 256, 0, stream>>>(X, g_attn, Xn, D);

  // 2-4) Q, K, V projections (f16 out)
  dim3 gP(D / BN, M / BM);
  gemm_f16_kernel<1><<<gP, 256, 0, stream>>>(Xn, Wq, nullptr, Qb, nullptr, M, D, D);
  gemm_f16_kernel<1><<<gP, 256, 0, stream>>>(Xn, Wk, nullptr, Kb, nullptr, M, D, D);
  gemm_f16_kernel<1><<<gP, 256, 0, stream>>>(Xn, Wv, nullptr, Vb, nullptr, M, D, D);

  // 5) causal flash attention (f16 out, heads concatenated)
  dim3 gA(S / 64, Bz * 16);
  attn_kernel<<<gA, 128, 0, stream>>>(Qb, Kb, Vb, AO, S);

  // 6) h = X + AO @ Wo^T  (f32)
  gemm_f16_kernel<2><<<gP, 256, 0, stream>>>(AO, Wo, Hb, nullptr, X, M, D, D);

  // 7) hn = RMSNorm(h) * g_ffn (f16, reuses Xn buffer)
  rmsnorm_f16_kernel<<<M, 256, 0, stream>>>(Hb, g_ffn, Xn, D);

  // 8) F1 = GELU(hn @ W1^T) (f16)
  dim3 gF1(F / BN, M / BM);
  gemm_f16_kernel<3><<<gF1, 256, 0, stream>>>(Xn, W1, nullptr, F1, nullptr, M, F, D);

  // 9) out = h + F1 @ W2^T (f32)
  dim3 gF2(D / BN, M / BM);
  gemm_f16_kernel<2><<<gF2, 256, 0, stream>>>(F1, W2, out, nullptr, Hb, M, D, F);
}
